// BottomUpHTMM_12515534701385
// MI455X (gfx1250) — matrix operand order repaired and checked by reading, not compiled, not run on hardware
//
#include <hip/hip_runtime.h>
#include <math.h>

typedef __attribute__((ext_vector_type(16))) _Float16 v16h;
typedef __attribute__((ext_vector_type(8)))  _Float16 v8h;
typedef __attribute__((ext_vector_type(8)))  float    v8f;

#define CC      8
#define LLp     4
#define NG      16
#define MMsz    8192
#define NNODES  5461
#define NINT    1365      /* STARTS[6]: internal nodes are 0..1364 */
#define NLEAF   4096
#define BATCHN  4
#define PERNODE (CC*NG)   /* 128 floats per node */

/* ws float offsets */
#define OFF_LSEB   0
#define OFF_LOGA   128
#define OFF_ASP    4224
#define OFF_LOGSP  8320
#define OFF_SMSP   8384
#define OFF_SMPI   8448
#define OFF_LOGPI  8960
#define OFF_BFRAG  9472          /* 8192 f16 = 4096 floats */
#define OFF_BATCH  13568
/* per-batch arrays (floats) */
#define SZ_BETA   ((size_t)NNODES * PERNODE)   /* 699008 */
#define SZ_EPS    ((size_t)NNODES * PERNODE)
#define SZ_TB     ((size_t)NINT   * PERNODE)   /* 174720 */
#define SZ_DSCR   ((size_t)1024 * 32 * 8)      /* 262144: max tasks x lanes x v8f */
#define SZ_PACKA  ((size_t)87 * 8192 / 2)      /* 712704 halves = 356352 floats */
#define PB_STRIDE (SZ_BETA + SZ_EPS + SZ_TB + SZ_DSCR + SZ_PACKA)

/* packed-A per-level offsets in halves: tiles {1,1,1,4,16,64}, 8192 halves/tile */
__device__ __constant__ int c_lvlOffH[6] = {0, 8192, 16384, 24576, 57344, 188416};

/* ---------------- precompute: small softmaxes + packed B fragments ----------- */
__global__ void htmm_precompute(const float* __restrict__ A,
                                const float* __restrict__ Pi,
                                const float* __restrict__ SP,
                                float* __restrict__ ws) {
  float* logA  = ws + OFF_LOGA;
  float* ASP   = ws + OFF_ASP;
  float* logSP = ws + OFF_LOGSP;
  float* smSP  = ws + OFF_SMSP;
  float* smPi  = ws + OFF_SMPI;
  float* logPi = ws + OFF_LOGPI;
  _Float16* Bfrag = (_Float16*)(ws + OFF_BFRAG);

  int t = threadIdx.x;
  if (t < NG) {                         /* SP: softmax over L per g */
    int g = t;
    float m = -INFINITY;
    for (int j = 0; j < LLp; ++j) m = fmaxf(m, SP[j*NG + g]);
    float s = 0.f;
    for (int j = 0; j < LLp; ++j) s += expf(SP[j*NG + g] - m);
    float ls = m + logf(s);
    for (int j = 0; j < LLp; ++j) {
      float lv = SP[j*NG + g] - ls;
      logSP[j*NG + g] = lv;
      smSP[j*NG + g]  = expf(lv);
    }
  }
  if (t < LLp*NG) {                     /* Pi: softmax over C per (j,g) */
    int j = t >> 4, g = t & 15;
    float m = -INFINITY;
    for (int c = 0; c < CC; ++c) m = fmaxf(m, Pi[c*LLp*NG + j*NG + g]);
    float s = 0.f;
    for (int c = 0; c < CC; ++c) s += expf(Pi[c*LLp*NG + j*NG + g] - m);
    float ls = m + logf(s);
    for (int c = 0; c < CC; ++c) {
      float lv = Pi[c*LLp*NG + j*NG + g] - ls;
      logPi[c*LLp*NG + j*NG + g] = lv;
      smPi[c*LLp*NG + j*NG + g]  = expf(lv);
    }
  }
  __syncthreads();
  /* A: softmax over axis 0 (i) per column col = k*64 + j*16 + g (512 cols) */
  {
    int col = t;                        /* blockDim.x == 512 */
    int j = (col >> 4) & 3, g = col & 15;
    float m = -INFINITY;
    for (int i = 0; i < CC; ++i) m = fmaxf(m, A[i*512 + col]);
    float s = 0.f;
    for (int i = 0; i < CC; ++i) s += expf(A[i*512 + col] - m);
    float ls = m + logf(s);
    float sp = smSP[j*NG + g];
    for (int i = 0; i < CC; ++i) {
      float lv = A[i*512 + col] - ls;
      logA[i*512 + col] = lv;
      ASP[i*512 + col]  = expf(lv) * sp;   /* sm_A * sm_SP */
    }
  }
  __syncthreads();
  /* bake B fragments: [g][lane][h] f16 per ISA 32x16 16-bit B layout */
  {
    int g = t >> 5, lane = t & 31;
    int mrow = lane & 15, hi = lane >> 4;
    for (int h = 0; h < 16; ++h) {
      int K2 = (h & 7) + 8*hi + ((h >> 3) << 4);
      int j2 = K2 >> 3, k2 = K2 & 7;
      float bv = (mrow < CC) ? ASP[mrow*512 + k2*64 + j2*16 + g] : 0.f;
      Bfrag[(g*32 + lane)*16 + h] = (_Float16)bv;
    }
  }
}

/* ---------------- lse over M for B: one block per (c,g) ---------------- */
__global__ void htmm_lseB(const float* __restrict__ B, float* __restrict__ lseB) {
  __shared__ float red[256];
  int c = blockIdx.x >> 4, g = blockIdx.x & 15;
  const float* base = B + (size_t)c * MMsz * NG + g;
  int tid = threadIdx.x;
  float m = -INFINITY;
  for (int mi = tid; mi < MMsz; mi += 256) m = fmaxf(m, base[(size_t)mi * NG]);
  red[tid] = m; __syncthreads();
  for (int s = 128; s > 0; s >>= 1) {
    if (tid < s) red[tid] = fmaxf(red[tid], red[tid + s]);
    __syncthreads();
  }
  m = red[0]; __syncthreads();
  float sum = 0.f;
  for (int mi = tid; mi < MMsz; mi += 256) sum += expf(base[(size_t)mi * NG] - m);
  red[tid] = sum; __syncthreads();
  for (int s = 128; s > 0; s >>= 1) {
    if (tid < s) red[tid] += red[tid + s];
    __syncthreads();
  }
  if (tid == 0) lseB[c*NG + g] = m + logf(red[0]);
}

/* ---------------- full tree E-step + likelihood: one block per batch -------- */
__global__ __launch_bounds__(1024)
void htmm_tree(const int* __restrict__ labels,
               const float* __restrict__ B,
               float* __restrict__ ws,
               float* __restrict__ out) {
  __shared__ float    sASP[4096];
  __shared__ float    sTacc[4096];
  __shared__ uint32_t sBfU[4096];          /* 8192 f16 B fragments */
  __shared__ float    sAl[LLp*NG];
  __shared__ float    sBl[NG];
  __shared__ float    sPl[NG];

  const int b    = blockIdx.x;
  const int tid  = threadIdx.x;
  const int lane = tid & 31;
  const int wave = tid >> 5;
  const int starts[8] = {0, 1, 5, 21, 85, 341, 1365, 5461};
  const int nds[6]    = {1, 4, 16, 64, 256, 1024};

  const float* lseB  = ws + OFF_LSEB;
  const float* logA  = ws + OFF_LOGA;
  const float* ASPg  = ws + OFF_ASP;
  const float* logSP = ws + OFF_LOGSP;
  const float* smPi  = ws + OFF_SMPI;
  const float* logPi = ws + OFF_LOGPI;
  const uint32_t* BfragU = (const uint32_t*)(ws + OFF_BFRAG);

  float* base  = ws + OFF_BATCH + (size_t)b * PB_STRIDE;
  float* beta  = base;
  float* eps   = beta + SZ_BETA;
  float* tb    = eps  + SZ_EPS;
  float* dscr  = tb   + SZ_TB;
  _Float16* packA = (_Float16*)(dscr + SZ_DSCR);
  const int* lab_b = labels + (size_t)b * NNODES;
  const _Float16* sBf = (const _Float16*)sBfU;

  for (int t = tid; t < 4096; t += 1024) {
    sASP[t] = ASPg[t]; sTacc[t] = 0.f; sBfU[t] = BfragU[t];
  }
  if (tid < LLp*NG) sAl[tid] = 0.f;
  if (tid < NG) { sBl[tid] = 0.f; sPl[tid] = 0.f; }
  /* zero packed-A tiles for the partial levels d=0,1 (halves [0,16384) = 8192 dwords) */
  for (int t = tid; t < 8192; t += 1024) ((uint32_t*)packA)[t] = 0u;
  __syncthreads();

  /* ---- leaves: beta_leaf = normalize_c( smPi[c,pos] * smB[c,lab] ); scatter d=5 ---- */
  for (int t = tid; t < NLEAF * NG; t += 1024) {
    int leaf = t >> 4, g = t & 15;
    int node = NINT + leaf;
    int pos  = (node - 1) & 3;
    int lab  = lab_b[node];
    float v[CC], s = 0.f;
#pragma unroll
    for (int c = 0; c < CC; ++c) {
      float bv = expf(B[((size_t)c * MMsz + lab) * NG + g] - lseB[c*NG + g]);
      v[c] = smPi[c*LLp*NG + pos*NG + g] * bv;
      s += v[c];
    }
    float inv = 1.f / s;
    v8h hv;
#pragma unroll
    for (int c = 0; c < CC; ++c) {
      float bb = v[c] * inv;
      beta[(size_t)node * PERNODE + c*NG + g] = bb;
      hv[c] = (_Float16)bb;
    }
    /* scatter packed A-fragment for parents at level 5 */
    int p = leaf >> 2, j = leaf & 3;
    int tile = p >> 4, mrow = p & 15;
    int hi = j & 1, hbase = (j >> 1) * 8;
    *(v8h*)&packA[c_lvlOffH[5] + ((tile*NG + g)*32 + mrow + 16*hi)*16 + hbase] = hv;
  }
  __syncthreads();

  /* ---- upward sweep: WMMA t_beta[p,i,g] = sum_{k,j} ASP[i,k,j,g]*beta_ch ---- */
  for (int d = 5; d >= 0; --d) {
    const int nd = nds[d];
    const int pstart = starts[d];
    const int tiles = (nd + 15) >> 4;
    const int tasks = tiles * NG;
    const _Float16* pA = packA + c_lvlOffH[d];

    auto doTask = [&](int task) {
      int g = task & 15;
      v16h afrag = *(const v16h*)&pA[(task*32 + lane)*16];
      v16h bfrag = *(const v16h*)&sBf[(g*32 + lane)*16];
      v8f acc = {};
      acc = __builtin_amdgcn_wmma_f32_16x16x32_f16(false, afrag, false, bfrag,
                                                   (short)0, acc, false, false);
      *(v8f*)&dscr[(task*32 + lane)*8] = acc;     /* D tile, unconditional */
    };
    for (int task = wave; task < tasks; task += 64) {
      doTask(task);
      int task2 = task + 32;
      if (task2 < tasks) doTask(task2);
    }
    __syncthreads();

    /* beta_d = normalize_i( t_beta * smB[i,lab] ); copy tb; scatter level d-1 */
    for (int t = tid; t < nd * NG; t += 1024) {
      int p = t >> 4, g = t & 15;
      int node = pstart + p;
      int lab  = lab_b[node];
      int tile = p >> 4, M = p & 15;
      int dhi = M >> 3, dv = M & 7;
      float v[CC], s = 0.f;
#pragma unroll
      for (int i = 0; i < CC; ++i) {
        float tbv = dscr[((tile*NG + g)*32 + i + 16*dhi)*8 + dv];
        tb[(size_t)node * PERNODE + i*NG + g] = tbv;
        float bv = expf(B[((size_t)i * MMsz + lab) * NG + g] - lseB[i*NG + g]);
        v[i] = tbv * bv;
        s += v[i];
      }
      float inv = 1.f / s;
      v8h hv;
#pragma unroll
      for (int i = 0; i < CC; ++i) {
        float bb = v[i] * inv;
        beta[(size_t)node * PERNODE + i*NG + g] = bb;
        hv[i] = (_Float16)bb;
      }
      if (d >= 1) {                     /* this node is a child for level d-1 */
        int ci = node - starts[d];
        int pp = ci >> 2, j = ci & 3;
        int ptile = pp >> 4, mrow = pp & 15;
        int hi = j & 1, hbase = (j >> 1) * 8;
        *(v8h*)&packA[c_lvlOffH[d-1] + ((ptile*NG + g)*32 + mrow + 16*hi)*16 + hbase] = hv;
      }
    }
    __syncthreads();
  }

  /* ---- downward sweep: eps, T_acc ---- */
  if (tid < PERNODE) eps[tid] = beta[tid];   /* eps[root] = beta[root] */
  __syncthreads();
  for (int d = 0; d < 6; ++d) {
    const int nd = nds[d];
    const int pstart = starts[d], cstart = starts[d + 1];
    for (int t = tid; t < nd * NG; t += 1024) {
      int p = t >> 4, g = t & 15;
      int node = pstart + p;
      float r[CC];
#pragma unroll
      for (int i = 0; i < CC; ++i)
        r[i] = eps[(size_t)node * PERNODE + i*NG + g] /
               tb [(size_t)node * PERNODE + i*NG + g];
#pragma unroll
      for (int j = 0; j < LLp; ++j) {
        int child = cstart + p * LLp + j;
#pragma unroll
        for (int k = 0; k < CC; ++k) {
          float bc = beta[(size_t)child * PERNODE + k*NG + g];
          float ek = 0.f;
#pragma unroll
          for (int i = 0; i < CC; ++i) {
            float te = r[i] * sASP[i*512 + k*64 + j*16 + g] * bc;
            atomicAdd(&sTacc[i*512 + k*64 + j*16 + g], te);
            ek += te;
          }
          eps[(size_t)child * PERNODE + k*NG + g] = ek;
        }
      }
    }
    __syncthreads();
  }

  /* ---- likelihood reductions ---- */
  for (int t = tid; t < NNODES * NG; t += 1024) {        /* B_lhood (all levels) */
    int node = t >> 4, g = t & 15;
    int lab = lab_b[node];
    float s = 0.f;
#pragma unroll
    for (int c = 0; c < CC; ++c)
      s += eps[(size_t)node * PERNODE + c*NG + g] *
           (B[((size_t)c * MMsz + lab) * NG + g] - lseB[c*NG + g]);
    atomicAdd(&sBl[g], s);
  }
  for (int t = tid; t < NLEAF * NG; t += 1024) {         /* Pi_lhood (leaves) */
    int leaf = t >> 4, g = t & 15;
    int node = NINT + leaf;
    int pos  = (node - 1) & 3;
    float s = 0.f;
#pragma unroll
    for (int c = 0; c < CC; ++c)
      s += eps[(size_t)node * PERNODE + c*NG + g] * logPi[c*LLp*NG + pos*NG + g];
    atomicAdd(&sPl[g], s);
  }
  for (int t = tid; t < 4096; t += 1024) {               /* A_lhood[j,g] */
    int j = (t >> 4) & 3, g = t & 15;
    atomicAdd(&sAl[j*NG + g], sTacc[t] * logA[t]);
  }
  __syncthreads();

  /* ---- out[b,i,k,j,g] = -(A_l[j,g] + B_l[g] + Pi_l[g] + Tacc*logSP[j,g]) ---- */
  for (int t = tid; t < 4096; t += 1024) {
    int j = (t >> 4) & 3, g = t & 15;
    out[(size_t)b * 4096 + t] =
        -(sAl[j*NG + g] + sBl[g] + sPl[g] + sTacc[t] * logSP[j*NG + g]);
  }
}

extern "C" void kernel_launch(void* const* d_in, const int* in_sizes, int n_in,
                              void* d_out, int out_size, void* d_ws, size_t ws_size,
                              hipStream_t stream) {
  const int*   labels = (const int*)  d_in[0];
  const float* A      = (const float*)d_in[1];
  const float* B      = (const float*)d_in[2];
  const float* Pi     = (const float*)d_in[3];
  const float* SP     = (const float*)d_in[4];
  float* out = (float*)d_out;
  float* ws  = (float*)d_ws;

  htmm_precompute<<<1, 512, 0, stream>>>(A, Pi, SP, ws);
  htmm_lseB<<<CC * NG, 256, 0, stream>>>(B, ws + OFF_LSEB);
  htmm_tree<<<BATCHN, 1024, 0, stream>>>(labels, B, ws, out);
}
